// ConvLayer_56453050139435
// MI455X (gfx1250) — matrix operand order repaired and checked by reading, not compiled, not run on hardware
//
#include <hip/hip_runtime.h>
#include <math.h>

#ifndef __has_builtin
#define __has_builtin(x) 0
#endif
#if __has_builtin(__builtin_amdgcn_tensor_load_to_lds)
#define USE_TDM 1
#else
#define USE_TDM 0
#endif

// ---------------- CDNA5 WMMA types ----------------
typedef __attribute__((ext_vector_type(16))) _Float16 v16h;
typedef __attribute__((ext_vector_type(8)))  _Float16 v8h;
typedef __attribute__((ext_vector_type(8)))  float    v8f;

#define IN_CH   128
#define OUT_CH  256
#define HW_IN   256
#define HW_OUT  128
#define BATCH   16
#define KEFF    6                      // folded 4x4 FIR + 3x3 conv -> 6x6
#define KPC     (KEFF*KEFF)            // 36 taps per input channel
#define KTOT    (IN_CH*KPC)            // 4608 = GEMM K
#define BK      32                     // K per WMMA step (16x16x32 f16)
#define NKT     (KTOT/BK)              // 144 K-tiles
#define LDA     40                     // padded LDS row stride (halves)
#define LDB     40
#define PHROW   136                    // halves per phase row: 4 pad + 128 + 4 pad
#define CH_STRIDE_H ((size_t)HW_IN * 2 * PHROW)   // halves per (c) in xph

union FragU { v16h v; v8h h[2]; };
union B16x { uint4 q[2]; _Float16 h[16]; };
union PK2  { _Float16 h[2]; unsigned int u; };

// ---------------------------------------------------------------------------
// Kernel 1: fold the [1,3,3,1]/8 separable FIR into the 3x3 conv weights.
// K order is (ky,kx,c):  k = (ky*6+kx)*128 + c   (tile = 32 channels, fixed tap)
// ---------------------------------------------------------------------------
__global__ void prep_weights(const float* __restrict__ w,
                             _Float16* __restrict__ weff) {
    int t = blockIdx.x * blockDim.x + threadIdx.x;     // one thread per (o,c)
    if (t >= OUT_CH * IN_CH) return;
    int o = t / IN_CH;
    int c = t - o * IN_CH;

    const float wscale = 1.0f / sqrtf(9.0f * (float)IN_CH);
    const float f1[4]  = {0.125f, 0.375f, 0.375f, 0.125f};

    float wl[3][3];
    const float* wp = w + ((size_t)o * IN_CH + c) * 9;
    #pragma unroll
    for (int a = 0; a < 3; ++a)
        #pragma unroll
        for (int b = 0; b < 3; ++b)
            wl[a][b] = wp[a * 3 + b] * wscale;

    #pragma unroll
    for (int ky = 0; ky < KEFF; ++ky) {
        #pragma unroll
        for (int kx = 0; kx < KEFF; ++kx) {
            float s = 0.0f;
            #pragma unroll
            for (int a = 0; a < 3; ++a) {
                int u = ky - a;
                if (u < 0 || u > 3) continue;
                #pragma unroll
                for (int b = 0; b < 3; ++b) {
                    int v = kx - b;
                    if (v < 0 || v > 3) continue;
                    s += wl[a][b] * f1[u] * f1[v];
                }
            }
            weff[(size_t)o * KTOT + (ky * KEFF + kx) * IN_CH + c] = (_Float16)s;
        }
    }
}

// ---------------------------------------------------------------------------
// Kernel 2: polyphase f16 pre-conversion of x.
// xph[n][c][iy][phase][136]:  [4 zero][128 samples of parity 'phase'][4 zero]
// ---------------------------------------------------------------------------
__global__ void prep_x(const float* __restrict__ x, _Float16* __restrict__ xph) {
    int rowid = blockIdx.x * 4 + (threadIdx.x >> 6);     // (n,c,iy) flattened
    int j     = threadIdx.x & 63;                        // 4 floats per thread
    const float* src = x + (size_t)rowid * HW_IN + j * 4;
    float4 v = *(const float4*)src;

    _Float16* pe = xph + (size_t)rowid * (2 * PHROW);    // even phase row
    _Float16* po = pe + PHROW;                           // odd phase row
    PK2 pk;
    pk.h[0] = (_Float16)v.x; pk.h[1] = (_Float16)v.z;
    *(unsigned int*)(pe + 4 + 2 * j) = pk.u;
    pk.h[0] = (_Float16)v.y; pk.h[1] = (_Float16)v.w;
    *(unsigned int*)(po + 4 + 2 * j) = pk.u;

    if (j == 0) {                                        // zero the pads
        *(unsigned long long*)(pe)       = 0ull;
        *(unsigned long long*)(pe + 132) = 0ull;
        *(unsigned long long*)(po)       = 0ull;
        *(unsigned long long*)(po + 132) = 0ull;
    }
}

#if USE_TDM
// ---------------------------------------------------------------------------
// TDM descriptor issue: copy 128 rows x 64B (stride 9216B) from w_eff into
// LDS with 16B padding after each row (-> LDA=40 halves). ISA 8.3/8.4 D#.
// amdgpu-toolchain builtin arity: (v4u, v8i, v4i, v4i, v8i, i32 cpol).
// ---------------------------------------------------------------------------
__device__ __forceinline__ void tdm_load_A(const _Float16* gsrc,
                                           unsigned lds_byte_off) {
    typedef unsigned int v4u_ __attribute__((ext_vector_type(4)));
    typedef int          v8i_ __attribute__((ext_vector_type(8)));
    typedef int          v4i_ __attribute__((ext_vector_type(4)));
    unsigned long long ga = (unsigned long long)(uintptr_t)gsrc;
    const unsigned tdim = 1u << 20;                    // large: never clips
    v4u_ g0;
    g0[0] = 1u;                                        // count=1, user mode
    g0[1] = lds_byte_off;                              // lds_addr
    g0[2] = (unsigned)(ga & 0xffffffffu);              // global_addr[31:0]
    g0[3] = (unsigned)((ga >> 32) & 0x1ffffffu)        // global_addr[56:32]
          | (2u << 30);                                // type = 2 ("image")
    v8i_ g1;
    g1[0] = (1 << 16)                                  // data_size = 2B
          | (1 << 20)                                  // pad_enable
          | (3 << 22)                                  // pad_interval: 16 DW (64B row)
          | (3 << 25);                                 // pad_amount: 4 DW (16B)
    g1[1] = (int)((tdim & 0xffffu) << 16);             // tensor_dim0[15:0]
    g1[2] = (int)((tdim >> 16) | ((tdim & 0xffffu) << 16)); // tdim0 hi | tdim1 lo
    g1[3] = (int)((tdim >> 16) | (32u << 16));         // tdim1 hi | tile_dim0=32
    g1[4] = 128;                                       // tile_dim1=128, tile_dim2=0
    g1[5] = KTOT;                                      // tensor_dim0_stride low32
    g1[6] = 0;
    g1[7] = 0;
    v4i_ gz4 = (v4i_){0, 0, 0, 0};                     // groups 2/3: 2D tensor
    v8i_ gz8 = (v8i_){0, 0, 0, 0, 0, 0, 0, 0};
    __builtin_amdgcn_tensor_load_to_lds(g0, g1, gz4, gz4, gz8, 0);
}
#endif

// ---------------------------------------------------------------------------
// Kernel 3: implicit-GEMM 6x6 stride-2 conv via v_wmma_f32_16x16x32_f16.
// Double-buffered LDS, one barrier/iteration; A-tile via TDM async copy.
// Grid: (2 o-tiles, 128 y, 16 n). Block: 256 threads = 8 wave32.
// ---------------------------------------------------------------------------
__global__ __launch_bounds__(256)
void conv_down_wmma(const _Float16* __restrict__ xph,
                    const _Float16* __restrict__ weff,
                    const float*    __restrict__ bias,
                    float*          __restrict__ out) {
    __shared__ __align__(16) _Float16 As[2][128 * LDA];
    __shared__ __align__(16) _Float16 Bs[2][128 * LDB];   // transposed: [pixel][k]

    const int o_base = blockIdx.x * 128;
    const int y      = blockIdx.y;
    const int n      = blockIdx.z;
    const int tid    = threadIdx.x;
    const int lane   = tid & 31;
    const int wave   = tid >> 5;
    const int wave_m = wave >> 2;           // 0..1  (64-row strips)
    const int wave_n = wave & 3;            // 0..3  (32-col strips)
    const int m_lane = lane & 15;
    const int kh     = lane >> 4;           // K-half held by this lane

    v8f acc[4][2];
    #pragma unroll
    for (int i = 0; i < 4; ++i)
        #pragma unroll
        for (int j = 0; j < 2; ++j)
            acc[i][j] = (v8f){0,0,0,0,0,0,0,0};

    // B staging: thread -> (channel pair, 8-pixel segment)
    const int cpair = tid & 15;             // channels 2*cpair, 2*cpair+1
    const int seg8  = tid >> 4;             // 0..15 -> pixels [8*seg8, 8*seg8+8)
    const size_t nbase = (size_t)n * IN_CH;
    B16x r0, r1;                            // staged B windows (two channels)

#if !USE_TDM
    // Fallback A staging: 2 threads per o-row, 16 halves each
    const int arow = tid >> 1;
    const int aseg = tid & 1;
    const _Float16* aptr = weff + (size_t)(o_base + arow) * KTOT + aseg * 16;
    uint4 a0, a1;
    auto loadA = [&](int kt) {
        a0 = *(const uint4*)(aptr + kt * BK);
        a1 = *(const uint4*)(aptr + kt * BK + 8);
        if (kt + 2 < NKT)
            __builtin_prefetch(aptr + kt * BK + 2 * BK, 0, 1);
    };
    auto storeA = [&](int buf) {
        *(uint4*)(&As[buf][arow * LDA + aseg * 16    ]) = a0;
        *(uint4*)(&As[buf][arow * LDA + aseg * 16 + 8]) = a1;
    };
#else
    const unsigned ldsA[2] = { (unsigned)(uintptr_t)(void*)&As[0][0],
                               (unsigned)(uintptr_t)(void*)&As[1][0] };
    auto issueA = [&](int buf, int kt) {
        if (wave == 0)
            tdm_load_A(weff + (size_t)o_base * KTOT + kt * BK, ldsA[buf]);
    };
#endif

    auto loadB = [&](int kt) {
        const int kyx = kt >> 2;
        const int cb  = (kt & 3) << 5;
        const int ky  = kyx / KEFF;
        const int kx  = kyx - ky * KEFF;
        const int p   = kx & 1;
        const int iy  = 2 * y + ky - 2;
        if (iy >= 0 && iy < HW_IN) {
            const _Float16* s0 = xph
                + ((nbase + (cb + 2 * cpair)) * HW_IN + iy) * (2 * PHROW)
                + (size_t)p * PHROW + seg8 * 8;
            const _Float16* s1 = s0 + CH_STRIDE_H;
            r0.q[0] = *(const uint4*)(s0);
            r0.q[1] = *(const uint4*)(s0 + 8);
            r1.q[0] = *(const uint4*)(s1);
            r1.q[1] = *(const uint4*)(s1 + 8);
            if ((kt & 3) != 3)
                __builtin_prefetch(s0 + 32 * CH_STRIDE_H, 0, 1);
        }
    };

    auto storeB = [&](int buf, int kt) {
        const int kyx = kt >> 2;
        const int ky  = kyx / KEFF;
        const int kx  = kyx - ky * KEFF;
        const int off = 3 + (kx >> 1);          // window start within 16 halves
        const int iy  = 2 * y + ky - 2;
        _Float16* bd = &Bs[buf][(seg8 * 8) * LDB + 2 * cpair];
        if (!(iy >= 0 && iy < HW_IN)) {
            #pragma unroll
            for (int j = 0; j < 8; ++j) *(unsigned int*)(&bd[j * LDB]) = 0u;
        } else if (off == 3) {
            #pragma unroll
            for (int j = 0; j < 8; ++j) {
                PK2 pk; pk.h[0] = r0.h[3 + j]; pk.h[1] = r1.h[3 + j];
                *(unsigned int*)(&bd[j * LDB]) = pk.u;
            }
        } else if (off == 4) {
            #pragma unroll
            for (int j = 0; j < 8; ++j) {
                PK2 pk; pk.h[0] = r0.h[4 + j]; pk.h[1] = r1.h[4 + j];
                *(unsigned int*)(&bd[j * LDB]) = pk.u;
            }
        } else {
            #pragma unroll
            for (int j = 0; j < 8; ++j) {
                PK2 pk; pk.h[0] = r0.h[5 + j]; pk.h[1] = r1.h[5 + j];
                *(unsigned int*)(&bd[j * LDB]) = pk.u;
            }
        }
    };

    auto compute = [&](int buf) {
        FragU a[4], b[2];
        #pragma unroll
        for (int mi = 0; mi < 4; ++mi) {
            int row = wave_m * 64 + mi * 16 + m_lane;
            a[mi].h[0] = *(const v8h*)(&As[buf][row * LDA +      kh * 8]);
            a[mi].h[1] = *(const v8h*)(&As[buf][row * LDA + 16 + kh * 8]);
        }
        #pragma unroll
        for (int ni = 0; ni < 2; ++ni) {
            int col = wave_n * 32 + ni * 16 + m_lane;
            b[ni].h[0] = *(const v8h*)(&Bs[buf][col * LDB +      kh * 8]);
            b[ni].h[1] = *(const v8h*)(&Bs[buf][col * LDB + 16 + kh * 8]);
        }
        #pragma unroll
        for (int mi = 0; mi < 4; ++mi)
            #pragma unroll
            for (int ni = 0; ni < 2; ++ni)
                acc[mi][ni] = __builtin_amdgcn_wmma_f32_16x16x32_f16(
                    false, a[mi].v, false, b[ni].v,
                    (short)0, acc[mi][ni], false, false);
    };

    // ---- software pipeline: stage tile 0 ----
#if USE_TDM
    issueA(0, 0);
#else
    loadA(0);
#endif
    loadB(0);
    storeB(0, 0);
#if USE_TDM
    __builtin_amdgcn_s_wait_tensorcnt(0);
#else
    storeA(0);
#endif
    __syncthreads();

    for (int kt = 0; kt < NKT; ++kt) {
        const int cur  = kt & 1;
        const int nxt  = cur ^ 1;
        const bool more = (kt + 1) < NKT;
        if (more) {
#if USE_TDM
            issueA(nxt, kt + 1);        // async TDM copy under the WMMAs
#else
            loadA(kt + 1);
#endif
            loadB(kt + 1);
        }
        compute(cur);
        if (more) {
            storeB(nxt, kt + 1);
#if !USE_TDM
            storeA(nxt);
#endif
        }
#if USE_TDM
        __builtin_amdgcn_s_wait_tensorcnt(0);
#endif
        __syncthreads();
    }

    // ---- epilogue: bias + leaky-ReLU*sqrt(2) + clamp, fp32 NCHW store ----
    const float slope  = 0.2f;
    const float gain   = 1.41421356237309515f;
    const float clampv = 256.0f;
    #pragma unroll
    for (int mi = 0; mi < 4; ++mi) {
        #pragma unroll
        for (int ni = 0; ni < 2; ++ni) {
            #pragma unroll
            for (int v = 0; v < 8; ++v) {
                int o  = o_base + wave_m * 64 + mi * 16 + v + 8 * kh;
                int xq = wave_n * 32 + ni * 16 + m_lane;
                float val = acc[mi][ni][v] + bias[o];
                val = (val >= 0.0f ? val : val * slope) * gain;
                val = fminf(fmaxf(val, -clampv), clampv);
                out[(((size_t)n * OUT_CH + o) * HW_OUT + y) * HW_OUT + xq] = val;
            }
        }
    }
}

// ---------------------------------------------------------------------------
extern "C" void kernel_launch(void* const* d_in, const int* in_sizes, int n_in,
                              void* d_out, int out_size, void* d_ws, size_t ws_size,
                              hipStream_t stream) {
    (void)in_sizes; (void)n_in; (void)out_size; (void)ws_size;
    const float* x    = (const float*)d_in[0];   // (16,128,256,256) f32
    const float* w    = (const float*)d_in[1];   // (256,128,3,3)    f32
    const float* bias = (const float*)d_in[2];   // (256,)           f32
    float*       out  = (float*)d_out;           // (16,256,128,128) f32

    // workspace layout: [ w_eff : 2.25 MB ][ xph : 272 MB ]
    _Float16* weff = (_Float16*)d_ws;
    _Float16* xph  = (_Float16*)((char*)d_ws + (size_t)OUT_CH * KTOT * sizeof(_Float16));

    prep_weights<<<(OUT_CH * IN_CH + 255) / 256, 256, 0, stream>>>(w, weff);

    const int nrows = BATCH * IN_CH * HW_IN;     // 524288 input rows
    prep_x<<<nrows / 4, 256, 0, stream>>>(x, xph);

    conv_down_wmma<<<dim3(2, HW_OUT, BATCH), 256, 0, stream>>>(xph, weff, bias, out);
}
